// SASRec_55851754717766
// MI455X (gfx1250) — compile-verified, matrix operand order checked by reading, not run on hardware
//
#include <hip/hip_runtime.h>

typedef __attribute__((ext_vector_type(16))) _Float16 v16h;
typedef __attribute__((ext_vector_type(8)))  _Float16 v8h;
typedef __attribute__((ext_vector_type(8)))  float    v8f;
typedef __attribute__((ext_vector_type(4)))  float    v4f;

#define B_   128
#define L_   200
#define D_   512
#define H_   8
#define HD_  64
#define NB_  2
#define M_   (B_ * L_)            // 25600 rows
#define QT_  13                   // ceil(200/16) query tiles
#define KPAD 224                  // keys padded to 7*32
#define EMB_SCALE 22.62741699796952f   // sqrt(512)
#define INV_SCALE 0.125f               // 1/sqrt(64)
#define LN_EPS 1e-8f
#define WSTRIDE 40                // LDS row stride (halfs) for 32-wide k slice, padded

// ---------------- helpers ----------------
__device__ inline v16h pack16(v8h lo, v8h hi) {
  v16h r;
#pragma unroll
  for (int e = 0; e < 8; ++e) { r[e] = lo[e]; r[e + 8] = hi[e]; }
  return r;
}

// load 16 f16 A/B-fragment elements from f16 row: k = [0..7] and [16..23] rel. ptr
__device__ inline v16h frag_h(const _Float16* p) {
  v8h lo = *(const v8h*)(p);
  v8h hi = *(const v8h*)(p + 16);
  return pack16(lo, hi);
}

// same but from f32 row with on-the-fly conversion
__device__ inline v16h frag_f(const float* p) {
  v4f a0 = *(const v4f*)(p);
  v4f a1 = *(const v4f*)(p + 4);
  v4f b0 = *(const v4f*)(p + 16);
  v4f b1 = *(const v4f*)(p + 20);
  v16h r;
#pragma unroll
  for (int e = 0; e < 4; ++e) {
    r[e]      = (_Float16)a0[e];
    r[e + 4]  = (_Float16)a1[e];
    r[e + 8]  = (_Float16)b0[e];
    r[e + 12] = (_Float16)b1[e];
  }
  return r;
}

// ---------------- embedding ----------------
__global__ void embed_kernel(const float* __restrict__ item, const float* __restrict__ e1,
                             const float* __restrict__ e2, const float* __restrict__ e3,
                             const float* __restrict__ pos,
                             const int* __restrict__ logs, const int* __restrict__ l1,
                             const int* __restrict__ l2, const int* __restrict__ l3,
                             float* __restrict__ seqs, float* __restrict__ side) {
  int gid = blockIdx.x * blockDim.x + threadIdx.x;       // over M_ * (D_/4)
  const int n4 = M_ * (D_ / 4);
  if (gid >= n4) return;
  int row = gid / (D_ / 4);
  int c4  = (gid % (D_ / 4)) * 4;
  int l   = row % L_;
  int it  = logs[row];
  int i1  = l1[row], i2 = l2[row], i3 = l3[row];
  v4f vs = *(const v4f*)(e1 + (size_t)i1 * D_ + c4);
  vs = vs + *(const v4f*)(e2 + (size_t)i2 * D_ + c4);
  vs = vs + *(const v4f*)(e3 + (size_t)i3 * D_ + c4);
  v4f vi = *(const v4f*)(item + (size_t)it * D_ + c4);
  v4f vp = *(const v4f*)(pos + (size_t)l * D_ + c4);
  float keep = (it != 0) ? 1.f : 0.f;
  v4f s  = ((vi + vs) * EMB_SCALE + vp) * keep;
  v4f ss = (vs * EMB_SCALE + vp) * keep;
  *(v4f*)(seqs + (size_t)row * D_ + c4) = s;
  *(v4f*)(side + (size_t)row * D_ + c4) = ss;
}

// ---------------- f32 -> f16 convert ----------------
__global__ void cvt_kernel(const float* __restrict__ src, _Float16* __restrict__ dst, int n) {
  int i = blockIdx.x * blockDim.x + threadIdx.x;
  if (i < n) dst[i] = (_Float16)src[i];
}

// ---------------- layernorm (block per row) ----------------
__global__ void ln_kernel(const float* __restrict__ x, const float* __restrict__ g,
                          const float* __restrict__ bta, float* __restrict__ out) {
  __shared__ float red[256];
  int row = blockIdx.x, tid = threadIdx.x;
  const float* xr = x + (size_t)row * D_;
  float v0 = xr[tid], v1 = xr[tid + 256];
  red[tid] = v0 + v1;
  __syncthreads();
#pragma unroll
  for (int s = 128; s > 0; s >>= 1) {
    if (tid < s) red[tid] += red[tid + s];
    __syncthreads();
  }
  float mean = red[0] * (1.f / D_);
  __syncthreads();
  float d0 = v0 - mean, d1 = v1 - mean;
  red[tid] = d0 * d0 + d1 * d1;
  __syncthreads();
#pragma unroll
  for (int s = 128; s > 0; s >>= 1) {
    if (tid < s) red[tid] += red[tid + s];
    __syncthreads();
  }
  float rstd = rsqrtf(red[0] * (1.f / D_) + LN_EPS);
  float* orow = out + (size_t)row * D_;
  orow[tid]       = d0 * rstd * g[tid]       + bta[tid];
  orow[tid + 256] = d1 * rstd * g[tid + 256] + bta[tid + 256];
}

// ---------------- WMMA GEMM: Y[M,N] = A[M,K] @ W[N,K]^T (+ epilogue) ----------------
// 8 waves / block; block tile 256Mx64N; W k-slice staged in LDS and shared by all
// waves (8x less L2 weight traffic); each wave computes 32 rows (8 WMMAs / k-step)
template <bool AH>
__global__ __launch_bounds__(256) void gemm_kernel(
    const void* __restrict__ A_, const _Float16* __restrict__ W,
    const float* __restrict__ bias,
    float* __restrict__ outA, const float* __restrict__ addA,
    float* __restrict__ outB, const float* __restrict__ addB,
    _Float16* __restrict__ out16, const int* __restrict__ keep,
    int relu, int N, int K) {
  __shared__ _Float16 Wl[64 * WSTRIDE];
  int tid  = threadIdx.x;
  int lane = tid & 31;
  int wave = tid >> 5;
  int mrow = lane & 15, half = lane >> 4;
  int m0 = blockIdx.x * 256 + wave * 32;
  int n0 = blockIdx.y * 64;
  v8f acc[2][4] = {};
  const float*    Af = (const float*)A_;
  const _Float16* Ah = (const _Float16*)A_;
  // W tile loader: each thread owns one 8-half chunk of the 64x32 slice
  int ln = tid >> 2;             // 0..63 (n within tile)
  int lc = (tid & 3) * 8;        // 0/8/16/24 (k within slice)
  const _Float16* Wg = W + (size_t)(n0 + ln) * K + lc;

  for (int k0 = 0; k0 < K; k0 += 32) {
    v8h wv = *(const v8h*)(Wg + k0);   // issue before barrier: overlaps the wait
    __syncthreads();                    // all waves done reading previous slice
    *(v8h*)(&Wl[ln * WSTRIDE + lc]) = wv;
    __syncthreads();

    int ka = k0 + half * 8;
    v16h a0, a1;
    if (AH) {
      a0 = frag_h(Ah + (size_t)(m0 + mrow) * K + ka);
      a1 = frag_h(Ah + (size_t)(m0 + 16 + mrow) * K + ka);
    } else {
      a0 = frag_f(Af + (size_t)(m0 + mrow) * K + ka);
      a1 = frag_f(Af + (size_t)(m0 + 16 + mrow) * K + ka);
    }
#pragma unroll
    for (int t = 0; t < 4; ++t) {
      const _Float16* bp = &Wl[(t * 16 + mrow) * WSTRIDE + half * 8];
      v16h b = pack16(*(const v8h*)(bp), *(const v8h*)(bp + 16));
      acc[0][t] = __builtin_amdgcn_wmma_f32_16x16x32_f16(false, a0, false, b, (short)0,
                                                         acc[0][t], false, false);
      acc[1][t] = __builtin_amdgcn_wmma_f32_16x16x32_f16(false, a1, false, b, (short)0,
                                                         acc[1][t], false, false);
    }
  }

#pragma unroll
  for (int r = 0; r < 2; ++r) {
#pragma unroll
    for (int t = 0; t < 4; ++t) {
#pragma unroll
      for (int j = 0; j < 8; ++j) {
        int row = m0 + r * 16 + j + half * 8;
        int col = n0 + t * 16 + mrow;
        float y = acc[r][t][j];
        if (bias) y += bias[col];
        if (relu) y = fmaxf(y, 0.f);
        float km = 1.f;
        if (keep) km = (keep[row] != 0) ? 1.f : 0.f;
        size_t idx = (size_t)row * N + col;
        if (outB) outB[idx] = (y + addB[idx]) * km;
        float ya = y + (addA ? addA[idx] : 0.f);
        ya *= km;
        if (outA) outA[idx] = ya;
        if (out16) out16[idx] = (_Float16)ya;
      }
    }
  }
}

// ---------------- fused dual-energy attention ----------------
// one wave per (b, h, 16-query tile); keys padded to 224
__global__ void attn_kernel(const _Float16* __restrict__ Q, const _Float16* __restrict__ K,
                            const _Float16* __restrict__ Qs, const _Float16* __restrict__ Ks,
                            const _Float16* __restrict__ V, _Float16* __restrict__ X) {
  __shared__ float     sc[16][KPAD];
  __shared__ _Float16  pr[16][KPAD];
  __shared__ _Float16  vl[KPAD][HD_];

  int bid = blockIdx.x;
  int qt = bid % QT_;
  int bh = bid / QT_;
  int b  = bh / H_;
  int h  = bh % H_;
  int lane = threadIdx.x;
  int m = lane & 15, half = lane >> 4;
  int q0 = qt * 16;
  size_t base = (size_t)b * L_ * D_ + (size_t)h * HD_;

  // stage V[224][64] into LDS (zero-pad keys >= 200)
  for (int i = lane; i < KPAD * 8; i += 32) {
    int row = i >> 3;
    int c = (i & 7) * 8;
    v8h v = {};
    if (row < L_) v = *(const v8h*)(V + base + (size_t)row * D_ + c);
    *(v8h*)(&vl[row][c]) = v;
  }

  // energies: e (item) and es (side) for 14 key tiles
  const v16h zf = {};
  for (int nt = 0; nt < KPAD / 16; ++nt) {
    int kk0 = nt * 16;
    v8f ae = {}, as_ = {};
#pragma unroll
    for (int ks = 0; ks < 2; ++ks) {
      int f0 = ks * 32 + half * 8;
      int qrow = q0 + m;
      int krow = kk0 + m;
      v16h aq  = (qrow < L_) ? frag_h(Q  + base + (size_t)qrow * D_ + f0) : zf;
      v16h aqs = (qrow < L_) ? frag_h(Qs + base + (size_t)qrow * D_ + f0) : zf;
      v16h bk  = (krow < L_) ? frag_h(K  + base + (size_t)krow * D_ + f0) : zf;
      v16h bks = (krow < L_) ? frag_h(Ks + base + (size_t)krow * D_ + f0) : zf;
      ae  = __builtin_amdgcn_wmma_f32_16x16x32_f16(false, aq,  false, bk,  (short)0, ae,  false, false);
      as_ = __builtin_amdgcn_wmma_f32_16x16x32_f16(false, aqs, false, bks, (short)0, as_, false, false);
    }
#pragma unroll
    for (int j = 0; j < 8; ++j) {
      int q  = q0 + j + half * 8;
      int kk = kk0 + m;
      float val;
      if (kk >= L_) val = -1e30f;
      else          val = ((kk > q) ? as_[j] : ae[j]) * INV_SCALE;
      sc[j + half * 8][kk] = val;
    }
  }
  __syncthreads();

  // row softmax (lanes 0..15 own one row each)
  if (lane < 16) {
    float mx = -1e30f;
    for (int k = 0; k < KPAD; ++k) mx = fmaxf(mx, sc[lane][k]);
    float s = 0.f;
    for (int k = 0; k < KPAD; ++k) s += __expf(sc[lane][k] - mx);
    float rs = 1.f / s;
    for (int k = 0; k < KPAD; ++k)
      pr[lane][k] = (_Float16)(__expf(sc[lane][k] - mx) * rs);
  }
  __syncthreads();

  // x = probs @ V  (M=16, K=224, N=64)
#pragma unroll
  for (int ft = 0; ft < 4; ++ft) {
    v8f acc = {};
    for (int ks = 0; ks < KPAD / 32; ++ks) {
      int f0 = ks * 32 + half * 8;
      v16h ap = frag_h(&pr[m][f0]);
      v16h bv;
      int n = ft * 16 + m;
#pragma unroll
      for (int e = 0; e < 16; ++e) {
        int kk = ks * 32 + ((e >> 3) << 4) + (e & 7) + half * 8;
        bv[e] = vl[kk][n];
      }
      acc = __builtin_amdgcn_wmma_f32_16x16x32_f16(false, ap, false, bv, (short)0, acc, false, false);
    }
#pragma unroll
    for (int j = 0; j < 8; ++j) {
      int q = q0 + j + half * 8;
      if (q < L_) X[base + (size_t)q * D_ + ft * 16 + m] = (_Float16)acc[j];
    }
  }
}

// ---------------- final logits ----------------
__global__ void logits_kernel(const float* __restrict__ feats, const float* __restrict__ item,
                              const float* __restrict__ e1, const float* __restrict__ e2,
                              const float* __restrict__ e3,
                              const int* __restrict__ ps, const int* __restrict__ p1,
                              const int* __restrict__ p2, const int* __restrict__ p3,
                              const int* __restrict__ ns, const int* __restrict__ n1,
                              const int* __restrict__ n2, const int* __restrict__ n3,
                              float* __restrict__ out) {
  __shared__ float rp[128], rn[128];
  int row = blockIdx.x, tid = threadIdx.x;
  int c = tid * 4;
  v4f f = *(const v4f*)(feats + (size_t)row * D_ + c);
  v4f pe = *(const v4f*)(item + (size_t)ps[row] * D_ + c);
  pe = pe + *(const v4f*)(e1 + (size_t)p1[row] * D_ + c);
  pe = pe + *(const v4f*)(e2 + (size_t)p2[row] * D_ + c);
  pe = pe + *(const v4f*)(e3 + (size_t)p3[row] * D_ + c);
  v4f ne = *(const v4f*)(item + (size_t)ns[row] * D_ + c);
  ne = ne + *(const v4f*)(e1 + (size_t)n1[row] * D_ + c);
  ne = ne + *(const v4f*)(e2 + (size_t)n2[row] * D_ + c);
  ne = ne + *(const v4f*)(e3 + (size_t)n3[row] * D_ + c);
  float dp = f[0] * pe[0] + f[1] * pe[1] + f[2] * pe[2] + f[3] * pe[3];
  float dn = f[0] * ne[0] + f[1] * ne[1] + f[2] * ne[2] + f[3] * ne[3];
  rp[tid] = dp; rn[tid] = dn;
  __syncthreads();
#pragma unroll
  for (int s = 64; s > 0; s >>= 1) {
    if (tid < s) { rp[tid] += rp[tid + s]; rn[tid] += rn[tid + s]; }
    __syncthreads();
  }
  if (tid == 0) { out[row] = rp[0]; out[M_ + row] = rn[0]; }
}

// ---------------- host orchestration ----------------
extern "C" void kernel_launch(void* const* d_in, const int* in_sizes, int n_in,
                              void* d_out, int out_size, void* d_ws, size_t ws_size,
                              hipStream_t stream) {
  const float* item_emb = (const float*)d_in[0];
  const float* s1_emb   = (const float*)d_in[1];
  const float* s2_emb   = (const float*)d_in[2];
  const float* s3_emb   = (const float*)d_in[3];
  const float* pos_emb  = (const float*)d_in[4];
  const float* wgt[8];            // wq wk wqs wks wv wo w1 w2
  for (int i = 0; i < 8; ++i) wgt[i] = (const float*)d_in[5 + i];
  const float* bia[8];            // bq bk bqs bks bv bo b1 b2
  for (int i = 0; i < 8; ++i) bia[i] = (const float*)d_in[13 + i];
  const float* ln_attn_g = (const float*)d_in[21];
  const float* ln_attn_b = (const float*)d_in[22];
  const float* ln_fwd_g  = (const float*)d_in[23];
  const float* ln_fwd_b  = (const float*)d_in[24];
  const float* last_g    = (const float*)d_in[25];
  const float* last_b    = (const float*)d_in[26];
  const int* log_seqs = (const int*)d_in[28];
  const int* pos_seqs = (const int*)d_in[29];
  const int* neg_seqs = (const int*)d_in[30];
  const int* log_s1 = (const int*)d_in[31];
  const int* pos_s1 = (const int*)d_in[32];
  const int* neg_s1 = (const int*)d_in[33];
  const int* log_s2 = (const int*)d_in[34];
  const int* pos_s2 = (const int*)d_in[35];
  const int* neg_s2 = (const int*)d_in[36];
  const int* log_s3 = (const int*)d_in[37];
  const int* pos_s3 = (const int*)d_in[38];
  const int* neg_s3 = (const int*)d_in[39];
  (void)in_sizes; (void)n_in; (void)out_size; (void)ws_size;

  const size_t MD = (size_t)M_ * D_;
  unsigned char* p = (unsigned char*)d_ws;
  auto carve = [&](size_t bytes) {
    void* r = (void*)p;
    p += (bytes + 255) & ~(size_t)255;
    return r;
  };
  float* seqs = (float*)carve(MD * 4);
  float* side = (float*)carve(MD * 4);
  float* Qin  = (float*)carve(MD * 4);
  float* Qsin = (float*)carve(MD * 4);
  _Float16* Q16  = (_Float16*)carve(MD * 2);
  _Float16* K16  = (_Float16*)carve(MD * 2);
  _Float16* Qs16 = (_Float16*)carve(MD * 2);
  _Float16* Ks16 = (_Float16*)carve(MD * 2);
  _Float16* V16  = (_Float16*)carve(MD * 2);
  _Float16* x16  = (_Float16*)carve(MD * 2);
  _Float16* h16  = x16;  // FFN hidden reuses attention-output buffer (consumed before)
  _Float16* w16[8];
  const int WN = NB_ * D_ * D_;
  for (int i = 0; i < 8; ++i) w16[i] = (_Float16*)carve((size_t)WN * 2);

  // convert weights to f16 once
  for (int i = 0; i < 8; ++i)
    cvt_kernel<<<(WN + 255) / 256, 256, 0, stream>>>(wgt[i], w16[i], WN);

  // embeddings -> seqs / seqs_side
  {
    int n4 = M_ * (D_ / 4);
    embed_kernel<<<(n4 + 255) / 256, 256, 0, stream>>>(
        item_emb, s1_emb, s2_emb, s3_emb, pos_emb,
        log_seqs, log_s1, log_s2, log_s3, seqs, side);
  }

  dim3 ggrid(M_ / 256, D_ / 64);
  auto gemmF = [&](const float* A, const _Float16* W, const float* bias, float* oA,
                   const float* aA, float* oB, const float* aB, _Float16* o16,
                   const int* keep, int relu) {
    gemm_kernel<false><<<ggrid, 256, 0, stream>>>(A, W, bias, oA, aA, oB, aB, o16, keep,
                                                  relu, D_, D_);
  };
  auto gemmH = [&](const _Float16* A, const _Float16* W, const float* bias, float* oA,
                   const float* aA, float* oB, const float* aB, _Float16* o16,
                   const int* keep, int relu) {
    gemm_kernel<true><<<ggrid, 256, 0, stream>>>(A, W, bias, oA, aA, oB, aB, o16, keep,
                                                 relu, D_, D_);
  };

  const size_t WOF = (size_t)D_ * D_;
  for (int i = 0; i < NB_; ++i) {
    const float* ga = ln_attn_g + i * D_;
    const float* ba = ln_attn_b + i * D_;
    const float* gf = ln_fwd_g + i * D_;
    const float* bf = ln_fwd_b + i * D_;

    // pre-attention LN
    ln_kernel<<<M_, 256, 0, stream>>>(seqs, ga, ba, Qin);
    ln_kernel<<<M_, 256, 0, stream>>>(side, ga, ba, Qsin);

    // projections (f16 outputs for attention)
    gemmF(Qin,  w16[0] + i * WOF, bia[0] + i * D_, nullptr, nullptr, nullptr, nullptr, Q16,  nullptr, 0);
    gemmF(seqs, w16[1] + i * WOF, bia[1] + i * D_, nullptr, nullptr, nullptr, nullptr, K16,  nullptr, 0);
    gemmF(Qsin, w16[2] + i * WOF, bia[2] + i * D_, nullptr, nullptr, nullptr, nullptr, Qs16, nullptr, 0);
    gemmF(side, w16[3] + i * WOF, bia[3] + i * D_, nullptr, nullptr, nullptr, nullptr, Ks16, nullptr, 0);
    gemmF(seqs, w16[4] + i * WOF, bia[4] + i * D_, nullptr, nullptr, nullptr, nullptr, V16,  nullptr, 0);

    // fused dual-energy attention
    attn_kernel<<<B_ * H_ * QT_, 32, 0, stream>>>(Q16, K16, Qs16, Ks16, V16, x16);

    // output projection + dual residual: seqs = Qin + y ; side = Qsin + y
    gemmH(x16, w16[5] + i * WOF, bia[5] + i * D_, seqs, Qin, side, Qsin, nullptr, nullptr, 0);

    // post-attention LN (in place)
    ln_kernel<<<M_, 256, 0, stream>>>(seqs, gf, bf, seqs);
    ln_kernel<<<M_, 256, 0, stream>>>(side, gf, bf, side);

    // FFN for item stream: h = relu(seqs@w1+b1); seqs = (h@w2+b2+seqs)*keep
    gemmF(seqs, w16[6] + i * WOF, bia[6] + i * D_, nullptr, nullptr, nullptr, nullptr, h16, nullptr, 1);
    gemmH(h16,  w16[7] + i * WOF, bia[7] + i * D_, seqs, seqs, nullptr, nullptr, nullptr, log_seqs, 0);

    // FFN for side stream (reuses h16 after item FFN completes)
    gemmF(side, w16[6] + i * WOF, bia[6] + i * D_, nullptr, nullptr, nullptr, nullptr, h16, nullptr, 1);
    gemmH(h16,  w16[7] + i * WOF, bia[7] + i * D_, side, side, nullptr, nullptr, nullptr, log_seqs, 0);
  }

  // final LN -> log_feats (reuse Qin)
  ln_kernel<<<M_, 256, 0, stream>>>(seqs, last_g, last_b, Qin);

  // logits
  logits_kernel<<<M_, 128, 0, stream>>>(Qin, item_emb, s1_emb, s2_emb, s3_emb,
                                        pos_seqs, pos_s1, pos_s2, pos_s3,
                                        neg_seqs, neg_s1, neg_s2, neg_s3,
                                        (float*)d_out);
}
